// S4DLayer_38792144618289
// MI455X (gfx1250) — compile-verified, matrix operand order checked by reading, not run on hardware
//
#include <hip/hip_runtime.h>
#include <hip/hip_bf16.h>

// Problem constants (match reference)
#define BATCH 16
#define SEQL  2048
#define HCH   512
#define NST   64
#define LTILES 128            // SEQL / 16
#define SLAB_ELEMS 512        // 16x32 bf16 slab
#define MAXBT 64              // max B tiles staged per block (i<128 -> t<=63)

typedef __attribute__((ext_vector_type(16))) __bf16 v16bf;
typedef __attribute__((ext_vector_type(8)))  float  v8f;

union V16U { uint4 q[2]; v16bf v; };

// ---------------------------------------------------------------------------
// Kernel 1: K[h,l] = Re( sum_n (C*Bb)_{hn} * exp(dt_h*Lam_n)^l )
// grid (HCH, SEQL/256), block 256
// ---------------------------------------------------------------------------
__global__ __launch_bounds__(256) void s4d_kernelK(
    const float* __restrict__ Lr, const float* __restrict__ Li,
    const float* __restrict__ Br, const float* __restrict__ Bi,
    const float* __restrict__ Cr, const float* __restrict__ Ci,
    const float* __restrict__ log_dt, float* __restrict__ K)
{
    const int h = blockIdx.x;
    const int l = blockIdx.y * 256 + threadIdx.x;
    const float dt = __expf(log_dt[h]);
    const float fl = (float)l;

    float sum = 0.f;
    #pragma unroll 4
    for (int n = 0; n < NST; ++n) {
        const float lamr = Lr[n];
        const float lami = Li[n];
        const float br = Br[h * NST + n];
        const float bi = Bi[h * NST + n];
        const float cr = Cr[h * NST + n];
        const float ci = Ci[h * NST + n];

        const float ar = dt * lamr;           // Re(dt*Lam)
        const float ai = dt * lami;           // Im(dt*Lam)

        // Ab = exp(dt*Lam);  Bb = B * (Ab - 1) / (Lam + 1e-8)
        const float ea = __expf(ar);
        float sb, cb; __sincosf(ai, &sb, &cb);
        const float numr = ea * cb - 1.f, numi = ea * sb;
        const float denr = lamr + 1e-8f, deni = lami;
        const float inv = 1.f / (denr * denr + deni * deni);
        const float qr = (numr * denr + numi * deni) * inv;
        const float qi = (numi * denr - numr * deni) * inv;
        const float bbr = br * qr - bi * qi;
        const float bbi = br * qi + bi * qr;
        // w = C * Bb
        const float wr = cr * bbr - ci * bbi;
        const float wi = cr * bbi + ci * bbr;

        // exp(dt*Lam)^l = e^{ar*l} * (cos(ai*l) + i sin(ai*l)); reduce angle in f64
        const float mag = __expf(ar * fl);
        double th = (double)ai * (double)l;
        th = fmod(th, 6.283185307179586);
        float s, c; __sincosf((float)th, &s, &c);
        sum += mag * (wr * c - wi * s);
    }
    K[h * SEQL + l] = sum;
}

// ---------------------------------------------------------------------------
// Kernel 2: expand K into 16x32 Toeplitz slabs in WMMA A-operand lane order.
// slab(h, m) element [r, k] = K[h, m*16 + r - k] (0 for negative lag).
// Lane mapping (16-bit A 16x32): lane<16: row r=lane, k = 0..7,16..23
//                                lane>=16: row r=lane-16, k = 8..15,24..31
// Memory order: slab*512 + lane*16 + idx   (idx 0..15 -> two b128 per lane)
// ---------------------------------------------------------------------------
__global__ __launch_bounds__(256) void s4d_slabs(
    const float* __restrict__ K, __bf16* __restrict__ slabs)
{
    const int gid  = blockIdx.x * 256 + threadIdx.x;   // < HCH*LTILES*512
    const int e    = gid & 511;
    const int slab = gid >> 9;
    const int m    = slab & (LTILES - 1);
    const int h    = slab >> 7;
    const int lane = e >> 4;
    const int idx  = e & 15;
    const int half = lane >> 4;
    const int r    = lane & 15;
    const int k    = (idx & 7) + (half << 3) + ((idx >> 3) << 4);
    const int lag  = m * 16 + r - k;
    const float v  = (lag >= 0) ? K[h * SEQL + lag] : 0.f;
    slabs[gid] = (__bf16)v;
}

// ---------------------------------------------------------------------------
// Kernel 3: u (B,L,H) f32  ->  ub[h][b][l] bf16 (column-contiguous B operand)
// ---------------------------------------------------------------------------
__global__ __launch_bounds__(256) void s4d_uconv(
    const float* __restrict__ u, __bf16* __restrict__ ub)
{
    const int gid = blockIdx.x * 256 + threadIdx.x;    // < HCH*BATCH*SEQL
    const int l  = gid & (SEQL - 1);
    const int hb = gid >> 11;
    const int b  = hb & (BATCH - 1);
    const int h  = hb >> 4;
    ub[gid] = (__bf16)u[((size_t)b * SEQL + l) * HCH + h];
}

// ---------------------------------------------------------------------------
// Kernel 4: triangular Toeplitz GEMM.
//   - block = 8 waves, one channel h, 8 consecutive row tiles i
//   - B (U) tiles for the whole block staged once in LDS (shared by waves)
//   - main loop unrolled x2 with two independent accumulator chains
//   - fused skip connection y += D[h]*u in epilogue
// ---------------------------------------------------------------------------
__global__ __launch_bounds__(256) void s4d_gemm(
    const __bf16* __restrict__ slabs, const __bf16* __restrict__ ub,
    const float* __restrict__ u, const float* __restrict__ D,
    float* __restrict__ out)
{
    __shared__ __bf16 Bsh[MAXBT * SLAB_ELEMS];          // 64 KB

    const int tid   = threadIdx.x;
    const int h     = blockIdx.x >> 4;                  // 512 channels
    const int ibase = (blockIdx.x & 15) << 3;           // first row tile

    // ---- stage U tiles t = 0..(imax>>1) into LDS in operand-lane order ----
    const int ntiles = ((ibase + 7) >> 1) + 1;
    const int nchunk = ntiles << 6;                     // 64 x 16B chunks/tile
    for (int c = tid; c < nchunk; c += 256) {
        const int tile = c >> 6;
        const int lane = (c >> 1) & 31;
        const int ck   = c & 1;
        const int col  = lane & 15;
        const int half = lane >> 4;
        const uint4* src = (const uint4*)(ub +
            ((size_t)(h * BATCH + col)) * SEQL + tile * 32 + half * 8 + ck * 16);
        *(uint4*)(&Bsh[tile * SLAB_ELEMS + lane * 16 + ck * 8]) = *src;
    }
    __syncthreads();

    const int lane = tid & 31;
    const int wave = tid >> 5;
    const int i    = ibase | wave;                      // row tile 0..127

    const __bf16* __restrict__ abase =
        slabs + (size_t)h * LTILES * SLAB_ELEMS + lane * 16;
    const __bf16* __restrict__ bl = &Bsh[lane * 16];

    v8f acc0 = {}, acc1 = {};
    const int tmax = i >> 1;
    int t = 0;
    // paired iterations: two independent WMMA chains
    for (; t + 1 <= tmax; t += 2) {
        const int m0 = i - (t << 1);
        const int m1 = m0 - 2;
        V16U A0, A1, B0, B1;
        const uint4* ap0 = (const uint4*)(abase + (size_t)m0 * SLAB_ELEMS);
        const uint4* ap1 = (const uint4*)(abase + (size_t)m1 * SLAB_ELEMS);
        A0.q[0] = ap0[0]; A0.q[1] = ap0[1];
        A1.q[0] = ap1[0]; A1.q[1] = ap1[1];
        const uint4* bp0 = (const uint4*)(bl + (size_t)t * SLAB_ELEMS);
        const uint4* bp1 = (const uint4*)(bl + (size_t)(t + 1) * SLAB_ELEMS);
        B0.q[0] = bp0[0]; B0.q[1] = bp0[1];
        B1.q[0] = bp1[0]; B1.q[1] = bp1[1];
        // branchless prefetch of the A slab two iterations ahead
        const int pm = (m1 > 4) ? (m1 - 4) : 0;
        __builtin_prefetch(abase + (size_t)pm * SLAB_ELEMS, 0, 1);
        acc0 = __builtin_amdgcn_wmma_f32_16x16x32_bf16(
                   false, A0.v, false, B0.v, (short)0, acc0, false, false);
        acc1 = __builtin_amdgcn_wmma_f32_16x16x32_bf16(
                   false, A1.v, false, B1.v, (short)0, acc1, false, false);
    }
    if (t <= tmax) {                                    // wave-uniform tail
        const int m0 = i - (t << 1);
        V16U A0, B0;
        const uint4* ap0 = (const uint4*)(abase + (size_t)m0 * SLAB_ELEMS);
        A0.q[0] = ap0[0]; A0.q[1] = ap0[1];
        const uint4* bp0 = (const uint4*)(bl + (size_t)t * SLAB_ELEMS);
        B0.q[0] = bp0[0]; B0.q[1] = bp0[1];
        acc0 = __builtin_amdgcn_wmma_f32_16x16x32_bf16(
                   false, A0.v, false, B0.v, (short)0, acc0, false, false);
    }

    // D-matrix layout: VGPR v, lanes 0-15 -> row i*16+v, lanes 16-31 -> row +8
    const float d = D[h];
    const int   b = lane & 15;
    const int rowbase = i * 16 + (lane >> 4) * 8;
    #pragma unroll
    for (int v = 0; v < 8; ++v) {
        const int l = rowbase + v;
        const size_t idx = ((size_t)b * SEQL + l) * HCH + h;
        out[idx] = (acc0[v] + acc1[v]) + d * u[idx];
    }
}

// ---------------------------------------------------------------------------
// Host launcher
// ---------------------------------------------------------------------------
extern "C" void kernel_launch(void* const* d_in, const int* in_sizes, int n_in,
                              void* d_out, int out_size, void* d_ws, size_t ws_size,
                              hipStream_t stream) {
    const float* u      = (const float*)d_in[0];
    const float* Lr     = (const float*)d_in[1];
    const float* Li     = (const float*)d_in[2];
    const float* Br     = (const float*)d_in[3];
    const float* Bi     = (const float*)d_in[4];
    const float* Cr     = (const float*)d_in[5];
    const float* Ci     = (const float*)d_in[6];
    const float* log_dt = (const float*)d_in[7];
    const float* Dv     = (const float*)d_in[8];
    float* out = (float*)d_out;

    // workspace layout
    char* ws = (char*)d_ws;
    float*  Kbuf  = (float*)ws;                                   // 4 MiB
    __bf16* slabs = (__bf16*)(ws + (size_t)HCH * SEQL * 4);       // 64 MiB
    __bf16* ubuf  = (__bf16*)((char*)slabs +
                     (size_t)HCH * LTILES * SLAB_ELEMS * 2);      // 32 MiB

    // 1) build K (H x L)
    s4d_kernelK<<<dim3(HCH, SEQL / 256), 256, 0, stream>>>(
        Lr, Li, Br, Bi, Cr, Ci, log_dt, Kbuf);

    // 2) expand K into WMMA-ready Toeplitz slabs
    {
        const int total = HCH * LTILES * SLAB_ELEMS;              // 33.5M
        s4d_slabs<<<total / 256, 256, 0, stream>>>(Kbuf, slabs);
    }

    // 3) transpose/convert u to bf16 [h][b][l]
    {
        const int total = HCH * BATCH * SEQL;                     // 16.8M
        s4d_uconv<<<total / 256, 256, 0, stream>>>(u, ubuf);
    }

    // 4) triangular Toeplitz GEMM + fused skip
    s4d_gemm<<<HCH * 16, 256, 0, stream>>>(slabs, ubuf, u, Dv, out);
}
